// BCJRDetector_9242769622582
// MI455X (gfx1250) — matrix operand order repaired and checked by reading, not compiled, not run on hardware
//
#include <hip/hip_runtime.h>
#include <hip/hip_bf16.h>
#include <math.h>

// BCJR detector for MI455X (gfx1250, wave32), fused backward+forward.
// B=1024 sequences, T=2048 steps, S=16 trellis states, MEM=4, V=4 channel words.
//
// One wave32 = one tile of 16 batch elements x 16 states.
//   - state vector in WMMA C/D layout: VGPR r, lane l -> state (r + 8*(l>=16)), batch (l%16)
//   - trellis transition (alpha' = (Adj @ alpha) .* g) = ONE v_wmma_f32_16x16x32_f16
//     per step. K slots: state u sits at K=u (element u%8 of lane half u/8),
//     so the B operand is each lane's OWN 8 states packed to f16 -- zero shuffles.
//     The 0/1 adjacency is exact in f16; alpha rescaled every step (f16 range safe;
//     the exact scale cancels in all comparisons, so approximate v_rcp is enough).
//   - likelihood g = 2^(y*A_s + B_s + cn*y^2): per-state fma+add+v_exp (log2e and
//     the quadratic expansion pre-folded into constants; y^2 shared by 8 states).
//   - w[t] = g*beta stored per-tile column-major [t][b%16][s]: backward writes 2x
//     float4/lane (1 KB/wave contiguous), forward reads 4x float4/lane. Per-tile w
//     is 2 MB; 64 tiles = 128 MB, resident in the 192 MB L2.
//   - per-step cross-lane traffic: 1 ds_bpermute (backward) / 3 (forward).

typedef __attribute__((ext_vector_type(16))) _Float16 v16h;
typedef __attribute__((ext_vector_type(8)))  float    v8f;

#define B_TOT 1024
#define T_LEN 2048
#define TILES (B_TOT / 16)
#define LOG2E 1.4426950408889634f

__device__ __forceinline__ float lane_xor16(float x) {
  return __shfl_xor(x, 16, 32);
}

// B operand: elements 0..7 = this lane's own 8 states (f16), 8..15 = 0.
// Low lanes cover K=0..7 (states 0..7), high lanes cover K=8..15 (states 8..15).
__device__ __forceinline__ v16h make_b_operand(const float* aDv) {
  v16h bb;
#pragma unroll
  for (int e = 0; e < 8; ++e) bb[e] = (_Float16)aDv[e];
#pragma unroll
  for (int e = 8; e < 16; ++e) bb[e] = (_Float16)0.f;
  return bb;
}

__global__ __launch_bounds__(32) void bcjr_fused(
    const float* __restrict__ y, const float* __restrict__ h,
    const int* __restrict__ snr_p, float* __restrict__ ws,
    float* __restrict__ out) {
  const int lane = threadIdx.x & 31;
  const int hi   = lane >> 4;
  const int col  = lane & 15;                 // batch-in-tile; also A-matrix row (state s)
  const int tile = blockIdx.x;
  const int b    = tile * 16 + col;
  const int v    = b & 3;

  const float snr    = (float)snr_p[0];
  const float sigma  = __powf(10.0f, -snr * 0.1f);
  const float inv2s2 = 1.0f / (2.0f * sigma * sigma);
  const float cn     = -inv2s2 * LOG2E;       // g = 2^(cn*(y-sp)^2), expanded below

  // Per-lane expected channel outputs sp[s] and folded likelihood constants:
  //   log2(g[r]) = cn*y^2 + y*gA[r] + gB[r]
  float gA[8], gB[8];
  {
    const float h0 = h[v * 4 + 0], h1 = h[v * 4 + 1];
    const float h2 = h[v * 4 + 2], h3 = h[v * 4 + 3];
#pragma unroll
    for (int r = 0; r < 8; ++r) {
      const int s = r + 8 * hi;
      float spv = ((s >> 0) & 1) ? -h0 : h0;
      spv      += ((s >> 1) & 1) ? -h1 : h1;
      spv      += ((s >> 2) & 1) ? -h2 : h2;
      spv      += ((s >> 3) & 1) ? -h3 : h3;
      gA[r] = -2.0f * cn * spv;               // y * (2c*sp*log2e), c = -cn/log2e
      gB[r] = cn * spv * spv;                 // -c*sp^2*log2e
    }
  }

  // Adjacency operands with state u at K(u) = u (element u%8 of lane half u/8).
  // All nonzeros live in elements 0-7 of both lane halves (they cover exactly K 0..15).
  v16h a_bwd, a_fwd;
  {
    const int eb = (2 * col) & 15;            // backward row s: succs 2s%16, +1
    const int ef = col >> 1;                  // forward  row s: preds s>>1, (s>>1)+8
#pragma unroll
    for (int e = 0; e < 16; ++e) {
      a_bwd[e] = (e < 8 && (e + 8 * hi == eb || e + 8 * hi == eb + 1)) ? (_Float16)1.f : (_Float16)0.f;
      a_fwd[e] = (e < 8 && (e + 8 * hi == ef || e + 8 * hi == ef + 8)) ? (_Float16)1.f : (_Float16)0.f;
    }
  }

  const size_t wbase = (size_t)tile * T_LEN * 256;
  const size_t ybase = (size_t)b * T_LEN;

  // ------------------------------------------------------------------
  // Phase 1: backward.  beta_t = norm((beta_{t+1}[2s%16]+beta_{t+1}[2s%16+1])*g_t)
  // store w[t][col][s] = g_t * beta_t  (each lane: states 8*hi..8*hi+7, contiguous)
  // ------------------------------------------------------------------
  {
    float bD[8];
#pragma unroll
    for (int r = 0; r < 8; ++r) bD[r] = 0.f;
    bD[0] = (hi == 0) ? 1.f : 0.f;            // state 0 = 1
    v16h bop = make_b_operand(bD);

    for (int t = T_LEN - 1; t >= 0; --t) {
      const float yv = y[ybase + t];
      const float yq = cn * yv * yv;          // shared by all 8 states
      float g[8];
#pragma unroll
      for (int r = 0; r < 8; ++r)
        g[r] = __builtin_amdgcn_exp2f(__builtin_fmaf(yv, gA[r], gB[r]) + yq);

      v8f cz = {};
      v8f dm = __builtin_amdgcn_wmma_f32_16x16x32_f16(
          false, a_bwd, false, bop, (short)0, cz, false, false);

#pragma unroll
      for (int r = 0; r < 8; ++r) bD[r] = dm[r] * g[r];
      float p0 = bD[0] + bD[1], p1 = bD[2] + bD[3];
      float p2 = bD[4] + bD[5], p3 = bD[6] + bD[7];
      float sum = (p0 + p1) + (p2 + p3);      // tree: depth 3, not 8
      sum += lane_xor16(sum);
      const float rs = __builtin_amdgcn_rcpf(sum + 1e-37f);
#pragma unroll
      for (int r = 0; r < 8; ++r) bD[r] *= rs;

      float* wp = ws + wbase + (size_t)t * 256 + col * 16 + hi * 8;
      float4 w0, w1;
      w0.x = g[0] * bD[0]; w0.y = g[1] * bD[1]; w0.z = g[2] * bD[2]; w0.w = g[3] * bD[3];
      w1.x = g[4] * bD[4]; w1.y = g[5] * bD[5]; w1.z = g[6] * bD[6]; w1.w = g[7] * bD[7];
      *(float4*)(wp)     = w0;
      *(float4*)(wp + 4) = w1;

      bop = make_b_operand(bD);
    }
  }

  // ------------------------------------------------------------------
  // Phase 2: forward + decode.  (same-wave store->load of w is ordered by HW;
  //  each tile's 2 MB of w is L2-resident)
  //   up_t = sum_s alpha_t[s]*w_t[2s%16], down_t = ... [2s%16+1]
  //   out[b][t+3] = (up < down);  alpha_{t+1} = norm((Adj_f @ alpha_t)*g_t)
  // ------------------------------------------------------------------
  {
    float aD[8];
#pragma unroll
    for (int r = 0; r < 8; ++r) aD[r] = 0.f;
    aD[0] = (hi == 0) ? 1.f : 0.f;            // alpha[0]: state 0 = 1
    v16h bop = make_b_operand(aD);

    const size_t obase = (size_t)b * T_LEN;
    // out[:, 0:3] = 0 (MEM-1 zero prefix); duplicate lanes write identical data
    out[obase] = 0.f; out[obase + 1] = 0.f; out[obase + 2] = 0.f;

    for (int t = 0; t < T_LEN; ++t) {
      // full 16-state w column for batch b (same for both lane halves)
      const float* wc = ws + wbase + (size_t)t * 256 + col * 16;
      float wcol[16];
      *(float4*)(wcol + 0)  = ((const float4*)wc)[0];
      *(float4*)(wcol + 4)  = ((const float4*)wc)[1];
      *(float4*)(wcol + 8)  = ((const float4*)wc)[2];
      *(float4*)(wcol + 12) = ((const float4*)wc)[3];

      // up/down with current alpha (pre-update == reference alpha[t]);
      // per-lane term for state s = r+8*hi uses w[2r], w[2r+1] (independent of hi).
      float upp[4], dnp[4];
#pragma unroll
      for (int i = 0; i < 4; ++i) {
        upp[i] = aD[2 * i] * wcol[4 * i]     + aD[2 * i + 1] * wcol[4 * i + 2];
        dnp[i] = aD[2 * i] * wcol[4 * i + 1] + aD[2 * i + 1] * wcol[4 * i + 3];
      }
      float up = (upp[0] + upp[1]) + (upp[2] + upp[3]);
      float dn = (dnp[0] + dnp[1]) + (dnp[2] + dnp[3]);
      up += lane_xor16(up);
      dn += lane_xor16(dn);
      if (t + 3 < T_LEN)                       // uniform branch; dup lanes write same value
        out[obase + t + 3] = (up < dn) ? 1.0f : 0.0f;

      // alpha update through the matrix pipe
      const float yv = y[ybase + t];
      const float yq = cn * yv * yv;
      float g[8];
#pragma unroll
      for (int r = 0; r < 8; ++r)
        g[r] = __builtin_amdgcn_exp2f(__builtin_fmaf(yv, gA[r], gB[r]) + yq);

      v8f cz = {};
      v8f dm = __builtin_amdgcn_wmma_f32_16x16x32_f16(
          false, a_fwd, false, bop, (short)0, cz, false, false);
#pragma unroll
      for (int r = 0; r < 8; ++r) aD[r] = dm[r] * g[r];
      float p0 = aD[0] + aD[1], p1 = aD[2] + aD[3];
      float p2 = aD[4] + aD[5], p3 = aD[6] + aD[7];
      float sum = (p0 + p1) + (p2 + p3);
      sum += lane_xor16(sum);
      const float rs = __builtin_amdgcn_rcpf(sum + 1e-37f);
#pragma unroll
      for (int r = 0; r < 8; ++r) aD[r] *= rs;

      bop = make_b_operand(aD);
    }
  }
}

extern "C" void kernel_launch(void* const* d_in, const int* in_sizes, int n_in,
                              void* d_out, int out_size, void* d_ws, size_t ws_size,
                              hipStream_t stream) {
  const float* y   = (const float*)d_in[0];   // [1024, 2048] f32
  const float* h   = (const float*)d_in[1];   // [4, 4] f32
  const int*   snr = (const int*)d_in[2];     // scalar
  float* out = (float*)d_out;                 // [1024, 2048] f32
  float* w   = (float*)d_ws;                  // needs 64*2048*256*4 = 128 MB (L2-resident)

  dim3 grid(TILES), block(32);
  bcjr_fused<<<grid, block, 0, stream>>>(y, h, snr, w, out);
}
